// AttnDecoderLSTM_55722905698542
// MI455X (gfx1250) — matrix operand order repaired and checked by reading, not compiled, not run on hardware
//
#include <hip/hip_runtime.h>
#include <hip/hip_bf16.h>
#include <stdint.h>

// ---------------- types for CDNA5 WMMA / TDM ----------------
typedef __attribute__((ext_vector_type(16))) __bf16 v16bf;
typedef __attribute__((ext_vector_type(8)))  float  v8f;
typedef __attribute__((ext_vector_type(4)))  int    si4;
typedef __attribute__((ext_vector_type(8)))  int    si8;

#define TSEQ 512
#define EDIM 256
#define HDIM 256
#define GDIM 1024   // 4*H
#define ODIM 32000
#define KD   512    // 2*H

__device__ __forceinline__ unsigned short f2bf(float x){
    __bf16 h = (__bf16)x;
    return __builtin_bit_cast(unsigned short, h);
}
__device__ __forceinline__ __bf16 bfraw(unsigned short u){
    return __builtin_bit_cast(__bf16, u);
}
__device__ __forceinline__ float sigm(float x){ return 1.0f/(1.0f+expf(-x)); }

// ---- Tensor Data Mover: 2D tile load (64 rows x 32 f32, row stride 512 f32) -> LDS ----
// D# per CDNA5 ISA. Issued wave-uniform; tracked by TENSORcnt.
__device__ __forceinline__ void tdm_load_b_tile(const float* gsrc, unsigned lds_off)
{
    union { si4 v; unsigned u[4]; } g0;
    union { si8 v; unsigned u[8]; } g1;
    unsigned long long ga = (unsigned long long)(uintptr_t)gsrc;
    g0.u[0] = 1u;                                        // count=1, user descriptor
    g0.u[1] = lds_off;                                   // lds_addr (bytes)
    g0.u[2] = (unsigned)(ga & 0xFFFFFFFFu);              // global_addr[31:0]
    g0.u[3] = (unsigned)((ga >> 32) & 0x01FFFFFFu)       // global_addr[56:32]
            | (2u << 30);                                // type=2 ("image")
    g1.u[0] = (2u << 16);                                // wg_mask=0, data_size=2 (4B)
    g1.u[1] = (32u << 16);                               // tensor_dim0 = 32
    g1.u[2] = (64u << 16);                               // tensor_dim1 = 64
    g1.u[3] = (32u << 16);                               // tile_dim0 = 32
    g1.u[4] = 64u;                                       // tile_dim1 = 64, tile_dim2 = 0
    g1.u[5] = 512u;                                      // tensor_dim0_stride = 512
    g1.u[6] = 0u;
    g1.u[7] = 0u;
    asm volatile("tensor_load_to_lds %0, %1" :: "s"(g0.v), "s"(g1.v) : "memory");
}

// ---------------- kernel 1: gather + bf16 conversions + bias fold + barrier reset ---------
__global__ void k_prep(const float* __restrict__ emb_table, const int* __restrict__ target,
                       const float* __restrict__ w_ih_f, const float* __restrict__ w_ih_b,
                       const float* __restrict__ b_ih_f, const float* __restrict__ b_hh_f,
                       const float* __restrict__ b_ih_b, const float* __restrict__ b_hh_b,
                       unsigned short* __restrict__ emb_bf, unsigned short* __restrict__ wih_bf,
                       float* __restrict__ biasc, unsigned* __restrict__ bar)
{
    int idx = blockIdx.x*256 + threadIdx.x;
    if (idx == 0){ bar[0] = 0u; bar[1] = 0u; }
    if (idx < TSEQ*EDIM){
        int t = idx >> 8, e = idx & 255;
        int tok = (t == 0) ? 0 : target[t-1];
        emb_bf[idx] = f2bf(emb_table[(size_t)tok*EDIM + e]);
    } else if (idx < TSEQ*EDIM + 2*GDIM*EDIM){
        int i2 = idx - TSEQ*EDIM;
        int cell = i2 >> 18;                   // 1024*256 = 2^18
        int r    = i2 & (GDIM*EDIM - 1);
        const float* w = cell ? w_ih_b : w_ih_f;
        wih_bf[i2] = f2bf(w[r]);
    } else if (idx < TSEQ*EDIM + 2*GDIM*EDIM + 2*GDIM){
        int i3 = idx - (TSEQ*EDIM + 2*GDIM*EDIM);
        int cell = i3 >> 10, n = i3 & 1023;
        biasc[i3] = cell ? (b_ih_b[n] + b_hh_b[n]) : (b_ih_f[n] + b_hh_f[n]);
    }
}

// ---------------- kernel 2: x-side gate GEMM  xg[cell][t][1024] = emb @ W_ih.T + bias -----
__global__ void k_xgates(const unsigned short* __restrict__ emb_bf,
                         const unsigned short* __restrict__ wih_bf,
                         const float* __restrict__ biasc, float* __restrict__ xg)
{
    int wave = threadIdx.x >> 5, lane = threadIdx.x & 31;
    int wt = blockIdx.x*8 + wave;              // 4096 wave-tiles total
    int cell = wt >> 11;
    int tl   = wt & 2047;
    int mt = tl >> 6, nt = tl & 63;
    int lr = lane & 15, half = lane >> 4;
    int m = mt*16 + lr;
    int n = nt*16 + lr;
    const unsigned short* wb = wih_bf + (size_t)cell*GDIM*EDIM;

    v8f acc;
    #pragma unroll
    for (int e = 0; e < 8; e++) acc[e] = 0.0f;

    #pragma unroll
    for (int kk = 0; kk < EDIM; kk += 32){
        v16bf a, b;
        #pragma unroll
        for (int i = 0; i < 8; i++){
            int ka = kk + 2*i + 16*half;
            unsigned av = *reinterpret_cast<const unsigned*>(emb_bf + (size_t)m*EDIM + ka);
            a[2*i]   = bfraw((unsigned short)av);
            a[2*i+1] = bfraw((unsigned short)(av >> 16));
            int kb = kk + 2*i + 16*half;
            unsigned bv = *reinterpret_cast<const unsigned*>(wb + (size_t)n*EDIM + kb);
            b[2*i]   = bfraw((unsigned short)bv);
            b[2*i+1] = bfraw((unsigned short)(bv >> 16));
        }
        acc = __builtin_amdgcn_wmma_f32_16x16x32_bf16(false, a, false, b, (short)0, acc, false, false);
    }
    float bias = biasc[cell*GDIM + n];
    #pragma unroll
    for (int r = 0; r < 8; r++){
        int row = mt*16 + r + 8*half;
        xg[((size_t)cell*TSEQ + row)*GDIM + n] = acc[r] + bias;
    }
}

// ---------------- kernel 3: persistent LSTM scan --------------------------------------
#define NWG_TOT 16u
__global__ void __launch_bounds__(256,1) k_scan(const float* __restrict__ w_hh_f,
                                                const float* __restrict__ w_hh_b,
                                                const float* __restrict__ xg,
                                                float* __restrict__ s,
                                                float* __restrict__ outp,
                                                float* __restrict__ hglob,
                                                unsigned* __restrict__ bar)
{
    __shared__ float hs[HDIM];
    __shared__ float gbuf[128];
    int tid  = threadIdx.x;
    int cell = blockIdx.x >> 3, slice = blockIdx.x & 7;
    int j = tid >> 1, half = tid & 1;          // 2 threads per gate row, K split 128/128
    int gate = j >> 5, unit = slice*32 + (j & 31);
    int grow = gate*HDIM + unit;               // row in [0,1024) of W_hh
    const float* W = (cell ? w_hh_b : w_hh_f) + (size_t)grow*HDIM + half*128;

    float4 rw[32];
    #pragma unroll
    for (int q = 0; q < 32; q++) rw[q] = reinterpret_cast<const float4*>(W)[q];

    hs[tid] = 0.0f;
    float creg = 0.0f;
    __syncthreads();
    const float4* h4 = reinterpret_cast<const float4*>(hs) + half*32;

    for (int t = 0; t < TSEQ; t++){
        float xi = 0.f, xf = 0.f, xc = 0.f, xo = 0.f;
        if (tid < 32){
            size_t base = ((size_t)cell*TSEQ + t)*GDIM;
            int u = slice*32 + tid;
            xi = xg[base + u];        xf = xg[base + 256 + u];
            xc = xg[base + 512 + u];  xo = xg[base + 768 + u];
            if (t + 1 < TSEQ){        // prefetch next step's gate rows (global_prefetch_b8)
                __builtin_prefetch(&xg[base + GDIM + u], 0, 1);
                __builtin_prefetch(&xg[base + GDIM + 512 + u], 0, 1);
            }
        }
        float acc = 0.0f;
        #pragma unroll
        for (int q = 0; q < 32; q++){
            float4 hv = h4[q];
            acc += rw[q].x*hv.x + rw[q].y*hv.y + rw[q].z*hv.z + rw[q].w*hv.w;
        }
        acc += __shfl_xor(acc, 1);
        if (!half) gbuf[j] = acc;
        __syncthreads();
        if (tid < 32){
            int u = slice*32 + tid;
            float gi = sigm(xi + gbuf[tid]);
            float gf = sigm(xf + gbuf[32 + tid]);
            float gc = tanhf(xc + gbuf[64 + tid]);
            float go = sigm(xo + gbuf[96 + tid]);
            creg = gf*creg + gi*gc;
            float h = go * tanhf(creg);
            __hip_atomic_store(&hglob[cell*HDIM + u], h, __ATOMIC_RELAXED, __HIP_MEMORY_SCOPE_AGENT);
            outp[(size_t)t*KD + cell*HDIM + u] = h;            // outputs[t] = cat(hf,hb)
            if (cell == 0){                                    // s[t] = cat(hf,cf)
                s[(size_t)t*KD + u] = h;
                s[(size_t)t*KD + HDIM + u] = creg;
            }
        }
        __syncthreads();
        if (tid == 0){
            __threadfence();
            unsigned gen = __hip_atomic_load(&bar[1], __ATOMIC_RELAXED, __HIP_MEMORY_SCOPE_AGENT);
            unsigned arr = __hip_atomic_fetch_add(&bar[0], 1u, __ATOMIC_ACQ_REL, __HIP_MEMORY_SCOPE_AGENT);
            if (arr == NWG_TOT - 1u){
                __hip_atomic_store(&bar[0], 0u, __ATOMIC_RELAXED, __HIP_MEMORY_SCOPE_AGENT);
                __hip_atomic_fetch_add(&bar[1], 1u, __ATOMIC_ACQ_REL, __HIP_MEMORY_SCOPE_AGENT);
            } else {
                while (__hip_atomic_load(&bar[1], __ATOMIC_ACQUIRE, __HIP_MEMORY_SCOPE_AGENT) == gen)
                    __builtin_amdgcn_s_sleep(1);
            }
        }
        __syncthreads();
        hs[tid] = __hip_atomic_load(&hglob[cell*HDIM + tid], __ATOMIC_RELAXED, __HIP_MEMORY_SCOPE_AGENT);
        __syncthreads();
    }
}

// ---------------- kernel 4: s -> bf16 ---------------------------------------------------
__global__ void k_sconv(const float* __restrict__ s, unsigned short* __restrict__ s_bf)
{
    int idx = blockIdx.x*256 + threadIdx.x;
    if (idx < TSEQ*KD) s_bf[idx] = f2bf(s[idx]);
}

// ---------------- kernel 5: logits = s @ test_w.T + test_b (bf16 WMMA + TDM staging) -----
// WG covers all M=512 x N=64 so test_w streams from HBM exactly once. B tiles are pulled
// by the Tensor Data Mover into double-buffered LDS (overlapped with WMMA), converted
// fp32->bf16 by one cooperative LDS pass, then consumed as WMMA fragments.
__global__ void __launch_bounds__(256,1) k_logits(const unsigned short* __restrict__ s_bf,
                                                  const float* __restrict__ test_w,
                                                  const float* __restrict__ test_b,
                                                  float* __restrict__ out)
{
    __shared__ __align__(16) float    BsF[2][64*32];   // fp32 TDM landing buffers (2 x 8KB)
    __shared__ __align__(16) unsigned Bs[64*16];       // bf16-packed tile (4KB)
    int tid = threadIdx.x;
    int wave = tid >> 5, lane = tid & 31;
    int lr = lane & 15, half = lane >> 4;
    int nb = blockIdx.x * 64;

    unsigned bsf_base = (unsigned)(uintptr_t)(&BsF[0][0]);   // LDS byte offset (addr[31:0])

    v8f acc[4][4];
    #pragma unroll
    for (int mi = 0; mi < 4; mi++)
        #pragma unroll
        for (int ni = 0; ni < 4; ni++)
            #pragma unroll
            for (int e = 0; e < 8; e++) acc[mi][ni][e] = 0.0f;

    if (wave == 0)   // prologue: kick off first tile
        tdm_load_b_tile(test_w + (size_t)nb*KD, bsf_base);

    for (int ki = 0; ki < 16; ki++){
        int kk = ki * 32;
        int cur = ki & 1;
        if (wave == 0){
            if (ki + 1 < 16){  // pipeline: issue next tile before waiting on current
                tdm_load_b_tile(test_w + (size_t)nb*KD + (kk + 32), bsf_base + (unsigned)((ki+1)&1)*8192u);
                __builtin_amdgcn_s_wait_tensorcnt(1);   // in-order: current tile landed
            } else {
                __builtin_amdgcn_s_wait_tensorcnt(0);   // last tile: drain
            }
        }
        __syncthreads();
        // cooperative fp32 -> packed bf16 conversion (done once per WG, not per wave)
        #pragma unroll
        for (int q = 0; q < 4; q++){
            int flat = tid*4 + q;                    // 1024 u32 = 64 cols x 16 pairs
            float f0 = BsF[cur][flat*2 + 0];
            float f1 = BsF[cur][flat*2 + 1];
            Bs[flat] = (unsigned)f2bf(f0) | ((unsigned)f2bf(f1) << 16);
        }
        __syncthreads();
        v16bf af[4], bfv[4];
        #pragma unroll
        for (int ni = 0; ni < 4; ni++){
            int n = ni*16 + lr;
            #pragma unroll
            for (int jx = 0; jx < 8; jx++){
                int kb = 2*jx + 16*half;
                unsigned bv = Bs[n*16 + (kb >> 1)];
                bfv[ni][2*jx]   = bfraw((unsigned short)bv);
                bfv[ni][2*jx+1] = bfraw((unsigned short)(bv >> 16));
            }
        }
        #pragma unroll
        for (int mi = 0; mi < 4; mi++){
            int row = (wave*4 + mi)*16 + lr;
            #pragma unroll
            for (int i = 0; i < 8; i++){
                int ka = kk + 2*i + 16*half;
                unsigned av = *reinterpret_cast<const unsigned*>(s_bf + (size_t)row*KD + ka);
                af[mi][2*i]   = bfraw((unsigned short)av);
                af[mi][2*i+1] = bfraw((unsigned short)(av >> 16));
            }
        }
        #pragma unroll
        for (int mi = 0; mi < 4; mi++)
            #pragma unroll
            for (int ni = 0; ni < 4; ni++)
                acc[mi][ni] = __builtin_amdgcn_wmma_f32_16x16x32_bf16(
                    false, af[mi], false, bfv[ni], (short)0, acc[mi][ni], false, false);
        __syncthreads();   // protect Bs / next TDM target buffer
    }
    #pragma unroll
    for (int mi = 0; mi < 4; mi++){
        #pragma unroll
        for (int ni = 0; ni < 4; ni++){
            int col = nb + ni*16 + lr;
            float bias = test_b[col];
            #pragma unroll
            for (int r = 0; r < 8; r++){
                int row = (wave*4 + mi)*16 + r + 8*half;
                out[(size_t)row*ODIM + col] = acc[mi][ni][r] + bias;
            }
        }
    }
}

// ---------------- kernel 6: row-wise log_softmax over 32000 -----------------------------
__global__ void k_lsm(float* __restrict__ out)
{
    __shared__ float rm[256], rs[256];
    int tid = threadIdx.x;
    float* p = out + (size_t)blockIdx.x*ODIM;
    float m = -3.4e38f, ssum = 0.0f;
    for (int i = tid; i < ODIM; i += 256){
        float x = p[i];
        float mn = fmaxf(m, x);
        ssum = ssum*expf(m - mn) + expf(x - mn);
        m = mn;
    }
    rm[tid] = m; rs[tid] = ssum;
    __syncthreads();
    for (int off = 128; off > 0; off >>= 1){
        if (tid < off){
            float m2 = rm[tid+off], s2 = rs[tid+off];
            float mn = fmaxf(rm[tid], m2);
            rs[tid] = rs[tid]*expf(rm[tid]-mn) + s2*expf(m2-mn);
            rm[tid] = mn;
        }
        __syncthreads();
    }
    float lsm = rm[0] + logf(rs[0]);
    for (int i = tid; i < ODIM; i += 256) p[i] = p[i] - lsm;
}

// ---------------- host-side launcher ----------------------------------------------------
extern "C" void kernel_launch(void* const* d_in, const int* in_sizes, int n_in,
                              void* d_out, int out_size, void* d_ws, size_t ws_size,
                              hipStream_t stream)
{
    (void)in_sizes; (void)n_in; (void)out_size; (void)ws_size;
    const float* emb_table = (const float*)d_in[1];
    const float* w_ih_f = (const float*)d_in[2];
    const float* w_hh_f = (const float*)d_in[3];
    const float* b_ih_f = (const float*)d_in[4];
    const float* b_hh_f = (const float*)d_in[5];
    const float* w_ih_b = (const float*)d_in[6];
    const float* w_hh_b = (const float*)d_in[7];
    const float* b_ih_b = (const float*)d_in[8];
    const float* b_hh_b = (const float*)d_in[9];
    const float* test_w = (const float*)d_in[20];
    const float* test_b = (const float*)d_in[21];
    const int*   target = (const int*)d_in[22];

    char* ws = (char*)d_ws;
    float*          xg      = (float*)(ws + 0);                 // 2*512*1024 f32 = 4 MB
    float*          s       = (float*)(ws + 4194304);           // 512*512 f32   = 1 MB
    unsigned short* emb_bf  = (unsigned short*)(ws + 5242880);  // 512*256 bf16
    unsigned short* wih_bf  = (unsigned short*)(ws + 5505024);  // 2*1024*256 bf16 = 1 MB
    unsigned short* s_bf    = (unsigned short*)(ws + 6553600);  // 512*512 bf16
    float*          biasc   = (float*)(ws + 7077888);           // 2*1024 f32
    float*          hglob   = (float*)(ws + 7086080);           // 512 f32
    unsigned*       bar     = (unsigned*)(ws + 7088128);        // barrier counter + generation

    float* logits  = (float*)d_out;                             // [512, 32000]
    float* outputs = (float*)d_out + (size_t)TSEQ*ODIM;         // [512, 512]

    k_prep  <<<dim3(2568), dim3(256), 0, stream>>>(emb_table, target, w_ih_f, w_ih_b,
                                                   b_ih_f, b_hh_f, b_ih_b, b_hh_b,
                                                   emb_bf, wih_bf, biasc, bar);
    k_xgates<<<dim3(512),  dim3(256), 0, stream>>>(emb_bf, wih_bf, biasc, xg);
    k_scan  <<<dim3(16),   dim3(256), 0, stream>>>(w_hh_f, w_hh_b, xg, s, outputs, hglob, bar);
    k_sconv <<<dim3(1024), dim3(256), 0, stream>>>(s, s_bf);
    k_logits<<<dim3(500),  dim3(256), 0, stream>>>(s_bf, test_w, test_b, logits);
    k_lsm   <<<dim3(512),  dim3(256), 0, stream>>>(logits);
}
